// Transformer_Conv_9517647528431
// MI455X (gfx1250) — matrix operand order repaired and checked by reading, not compiled, not run on hardware
//
#include <hip/hip_runtime.h>

typedef __attribute__((ext_vector_type(16))) _Float16 v16h;
typedef __attribute__((ext_vector_type(8)))  _Float16 v8h;
typedef __attribute__((ext_vector_type(8)))  float    v8f;
typedef _Float16 h16;

#define DEPTH  4
#define DIM    384
#define HEADS  8
#define DHEAD  64
#define INNER  512
#define HID    96
#define BB     8
#define NSEQ   1024
#define NTOK   (BB * NSEQ)   /* 8192 */
#define SCALE  0.125f        /* 64^-0.5 */

// ---------------------------------------------------------------------------
// WMMA helper (f16 16x16x32, f32 accumulate)
// ---------------------------------------------------------------------------
__device__ __forceinline__ v8f wmma_f16(v16h a, v16h b, v8f c) {
  return __builtin_amdgcn_wmma_f32_16x16x32_f16(
      /*neg_a=*/false, a, /*neg_b=*/false, b,
      /*c_mod=*/(short)0, c, /*reuse_a=*/false, /*reuse_b=*/false);
}

// A fragment: 16x32 tile, row-major source, leading dim lda (elements).
// Lanes 0-15: row m=lane, K in {0..7,16..23}; lanes 16-31: K in {8..15,24..31}.
__device__ __forceinline__ v16h load_a_frag(const h16* base, int lda) {
  int l = threadIdx.x & 31;
  const h16* p = base + (l & 15) * lda + ((l < 16) ? 0 : 8);
  v16h a;
  ((v8h*)&a)[0] = *(const v8h*)(p);
  ((v8h*)&a)[1] = *(const v8h*)(p + 16);
  return a;
}

// B fragment: 32x16 tile from B^T stored row-major [N][K], ld=ldb.
// Lanes 0-15: col j=lane, K in {0..7,16..23}; lanes 16-31: K in {8..15,24..31}.
__device__ __forceinline__ v16h load_b_frag(const h16* base, int ldb) {
  int l = threadIdx.x & 31;
  const h16* p = base + (l & 15) * ldb + ((l < 16) ? 0 : 8);
  v16h b;
  ((v8h*)&b)[0] = *(const v8h*)(p);
  ((v8h*)&b)[1] = *(const v8h*)(p + 16);
  return b;
}

// ---------------------------------------------------------------------------
// Weight transpose + f16 convert: src f32 [K][N] -> dst f16 [N][K]
// ---------------------------------------------------------------------------
__global__ void transpose_f16_kernel(const float* __restrict__ src,
                                     h16* __restrict__ dst, int K, int N) {
  int idx = blockIdx.x * blockDim.x + threadIdx.x;
  if (idx >= K * N) return;
  int k = idx / N, n = idx - k * N;
  dst[(size_t)n * K + k] = (h16)src[idx];
}

// ---------------------------------------------------------------------------
// LayerNorm over DIM=384, writes f16. One block (128 threads) per row.
// ---------------------------------------------------------------------------
__global__ void __launch_bounds__(128) ln_kernel(const float* __restrict__ x,
                                                 const float* __restrict__ w,
                                                 const float* __restrict__ b,
                                                 h16* __restrict__ out) {
  int row = blockIdx.x;
  const float* xr = x + (size_t)row * DIM;
  int tid = threadIdx.x;
  float v0 = xr[tid], v1 = xr[tid + 128], v2 = xr[tid + 256];
  float s = v0 + v1 + v2;
  float ss = v0 * v0 + v1 * v1 + v2 * v2;
  #pragma unroll
  for (int m = 16; m >= 1; m >>= 1) {
    s  += __shfl_xor(s,  m);
    ss += __shfl_xor(ss, m);
  }
  __shared__ float sh[8];
  int wave = tid >> 5, lane = tid & 31;
  if (lane == 0) { sh[wave] = s; sh[4 + wave] = ss; }
  __syncthreads();
  s  = sh[0] + sh[1] + sh[2] + sh[3];
  ss = sh[4] + sh[5] + sh[6] + sh[7];
  float mu  = s * (1.0f / DIM);
  float var = ss * (1.0f / DIM) - mu * mu;
  float inv = rsqrtf(var + 1e-5f);
  h16* orow = out + (size_t)row * DIM;
  orow[tid]       = (h16)((v0 - mu) * inv * w[tid]       + b[tid]);
  orow[tid + 128] = (h16)((v1 - mu) * inv * w[tid + 128] + b[tid + 128]);
  orow[tid + 256] = (h16)((v2 - mu) * inv * w[tid + 256] + b[tid + 256]);
}

// ---------------------------------------------------------------------------
// Generic GEMM: C[M][N] = A(f16)[M][K] @ Bt(f16)[N][K]^T (+bias +residual)
// One wave per MTxNT output tile. grid = (N/NT, M/MT), block = 32.
// ---------------------------------------------------------------------------
__device__ __forceinline__ void store_tile_f32(v8f c, int row0, int col0,
                                               const float* bias,
                                               const float* residual,
                                               float* C, int ldc, int hasres) {
  int l = threadIdx.x & 31;
  int col = col0 + (l & 15);
  int rbase = row0 + ((l < 16) ? 0 : 8);
  float bv = bias[col];
  #pragma unroll
  for (int r = 0; r < 8; ++r) {
    size_t idx = (size_t)(rbase + r) * ldc + col;
    float v = c[r] + bv;
    if (hasres) v += residual[idx];
    C[idx] = v;
  }
}

template <int MT, int NT>
__global__ void __launch_bounds__(32) gemm_kernel(const h16* __restrict__ A,
                                                  const h16* __restrict__ Bt,
                                                  const float* __restrict__ bias,
                                                  const float* __restrict__ residual,
                                                  float* __restrict__ C,
                                                  int N, int K, int lda, int ldb,
                                                  int hasres) {
  constexpr int MR = MT / 16, NR = NT / 16;
  int nt = blockIdx.x * NT;
  int mt = blockIdx.y * MT;
  v8f acc[MR][NR] = {};
  for (int kk = 0; kk < K; kk += 32) {
    v16h a[MR], b[NR];
    #pragma unroll
    for (int i = 0; i < MR; ++i)
      a[i] = load_a_frag(A + (size_t)(mt + i * 16) * lda + kk, lda);
    #pragma unroll
    for (int j = 0; j < NR; ++j)
      b[j] = load_b_frag(Bt + (size_t)(nt + j * 16) * ldb + kk, ldb);
    #pragma unroll
    for (int i = 0; i < MR; ++i)
      #pragma unroll
      for (int j = 0; j < NR; ++j)
        acc[i][j] = wmma_f16(a[i], b[j], acc[i][j]);
  }
  #pragma unroll
  for (int i = 0; i < MR; ++i)
    #pragma unroll
    for (int j = 0; j < NR; ++j)
      store_tile_f32(acc[i][j], mt + i * 16, nt + j * 16, bias, residual, C, N, hasres);
}

// ---------------------------------------------------------------------------
// QKV GEMM (64x64 per wave): xn[8192][384] @ wqkv_t[1536][384]^T, scattering
// q[bh][n][64], k[bh][n][64], vt[bh][64][n] (all f16; v stored transposed).
// ---------------------------------------------------------------------------
__device__ __forceinline__ void store_qkv_tile(v8f c, int row0, int col0,
                                               h16* q, h16* k, h16* vt) {
  int l = threadIdx.x & 31;
  int col = col0 + (l & 15);
  int which = col >> 9;          // 0=q 1=k 2=v (tile never straddles)
  int rc = col & 511;
  int h = rc >> 6, d = rc & 63;
  int rbase = row0 + ((l < 16) ? 0 : 8);
  #pragma unroll
  for (int r = 0; r < 8; ++r) {
    int row = rbase + r;
    int b = row >> 10, n = row & 1023;
    int bh = (b << 3) + h;
    h16 v = (h16)c[r];
    if (which == 0)      q [((size_t)bh * NSEQ + n) * DHEAD + d] = v;
    else if (which == 1) k [((size_t)bh * NSEQ + n) * DHEAD + d] = v;
    else                 vt[((size_t)bh * DHEAD + d) * NSEQ + n] = v;
  }
}

__global__ void __launch_bounds__(32) gemm_qkv_kernel(const h16* __restrict__ A,
                                                      const h16* __restrict__ Bt,
                                                      h16* __restrict__ q,
                                                      h16* __restrict__ k,
                                                      h16* __restrict__ vt) {
  int nt = blockIdx.x << 6;   // 64-wide
  int mt = blockIdx.y << 6;   // 64-tall
  v8f acc[4][4] = {};
  for (int kk = 0; kk < DIM; kk += 32) {
    v16h a[4], b[4];
    #pragma unroll
    for (int i = 0; i < 4; ++i)
      a[i] = load_a_frag(A + (size_t)(mt + i * 16) * DIM + kk, DIM);
    #pragma unroll
    for (int j = 0; j < 4; ++j)
      b[j] = load_b_frag(Bt + (size_t)(nt + j * 16) * DIM + kk, DIM);
    #pragma unroll
    for (int i = 0; i < 4; ++i)
      #pragma unroll
      for (int j = 0; j < 4; ++j)
        acc[i][j] = wmma_f16(a[i], b[j], acc[i][j]);
  }
  #pragma unroll
  for (int i = 0; i < 4; ++i)
    #pragma unroll
    for (int j = 0; j < 4; ++j)
      store_qkv_tile(acc[i][j], mt + i * 16, nt + j * 16, q, k, vt);
}

// ---------------------------------------------------------------------------
// Attention: per (b,h) x 16-row tile. 4 waves; wave w owns cols [w*256,+256).
// S = Q K^T * scale -> softmax -> (+attn_last,f16) -> store attn_last ->
// P(f16, LDS) @ V^T -> aout[b][n][h*64+d] (f16).
// attn_last is f16: 134 MB total => mostly L2-resident on MI455X.
// ---------------------------------------------------------------------------
__global__ void __launch_bounds__(128) attn_kernel(const h16* __restrict__ qg,
                                                   const h16* __restrict__ kg,
                                                   const h16* __restrict__ vtg,
                                                   h16* __restrict__ attn_last,
                                                   h16* __restrict__ aout,
                                                   int use_prev) {
  int bh   = blockIdx.x >> 6;
  int row0 = (blockIdx.x & 63) << 4;
  const h16* q  = qg  + ((size_t)bh * NSEQ + row0) * DHEAD;
  const h16* kk = kg  + (size_t)bh * NSEQ * DHEAD;
  const h16* vt = vtg + (size_t)bh * DHEAD * NSEQ;
  h16* prev = attn_last + (size_t)bh * NSEQ * NSEQ + (size_t)row0 * NSEQ;

  int tid  = threadIdx.x;
  int wave = tid >> 5;
  int l    = tid & 31;
  int lcol = l & 15;
  int radd = (l < 16) ? 0 : 8;
  int jbase = wave << 8;

  __shared__ h16   atile[16][NSEQ];      // 32 KB
  __shared__ float red[4][16];
  __shared__ float osum[4][16 * DHEAD];  // 16 KB

  // ---- scores ----
  v16h a0 = load_a_frag(q, DHEAD);
  v16h a1 = load_a_frag(q + 32, DHEAD);
  v8f s[16];
  #pragma unroll
  for (int t = 0; t < 16; ++t) {
    int j0 = jbase + (t << 4);
    v8f c = {};
    v16h b0 = load_b_frag(kk + (size_t)j0 * DHEAD, DHEAD);
    c = wmma_f16(a0, b0, c);
    v16h b1 = load_b_frag(kk + (size_t)j0 * DHEAD + 32, DHEAD);
    c = wmma_f16(a1, b1, c);
    s[t] = c;
  }

  // ---- row max ----
  float mx[8];
  #pragma unroll
  for (int r = 0; r < 8; ++r) mx[r] = -3.0e38f;
  #pragma unroll
  for (int t = 0; t < 16; ++t) {
    #pragma unroll
    for (int r = 0; r < 8; ++r) {
      s[t][r] *= SCALE;
      mx[r] = fmaxf(mx[r], s[t][r]);
    }
  }
  #pragma unroll
  for (int m = 1; m < 16; m <<= 1) {
    #pragma unroll
    for (int r = 0; r < 8; ++r) mx[r] = fmaxf(mx[r], __shfl_xor(mx[r], m));
  }
  if (l == 0 || l == 16) {
    #pragma unroll
    for (int r = 0; r < 8; ++r) red[wave][radd + r] = mx[r];
  }
  __syncthreads();
  float rmax[8];
  #pragma unroll
  for (int r = 0; r < 8; ++r)
    rmax[r] = fmaxf(fmaxf(red[0][radd + r], red[1][radd + r]),
                    fmaxf(red[2][radd + r], red[3][radd + r]));
  __syncthreads();

  // ---- exp + row sum ----
  float sm[8];
  #pragma unroll
  for (int r = 0; r < 8; ++r) sm[r] = 0.0f;
  #pragma unroll
  for (int t = 0; t < 16; ++t) {
    #pragma unroll
    for (int r = 0; r < 8; ++r) {
      float e = expf(s[t][r] - rmax[r]);
      s[t][r] = e;
      sm[r] += e;
    }
  }
  #pragma unroll
  for (int m = 1; m < 16; m <<= 1) {
    #pragma unroll
    for (int r = 0; r < 8; ++r) sm[r] += __shfl_xor(sm[r], m);
  }
  if (l == 0 || l == 16) {
    #pragma unroll
    for (int r = 0; r < 8; ++r) red[wave][radd + r] = sm[r];
  }
  __syncthreads();
  float inv[8];
  #pragma unroll
  for (int r = 0; r < 8; ++r)
    inv[r] = 1.0f / (red[0][radd + r] + red[1][radd + r] +
                     red[2][radd + r] + red[3][radd + r]);

  // ---- finalize probs, cross-layer residual (f16), stash f16 tile ----
  #pragma unroll
  for (int t = 0; t < 16; ++t) {
    int j0 = jbase + (t << 4);
    #pragma unroll
    for (int r = 0; r < 8; ++r) {
      int row = radd + r;
      float v = s[t][r] * inv[r];
      size_t pidx = (size_t)row * NSEQ + j0 + lcol;
      if (use_prev) v += (float)prev[pidx];
      h16 hv = (h16)v;
      prev[pidx] = hv;
      atile[row][j0 + lcol] = hv;   // O-GEMM uses exactly the stored value
    }
  }
  __syncthreads();

  // ---- O = P @ V^T over this wave's 256-col K range ----
  v8f o[4] = {{}, {}, {}, {}};
  #pragma unroll
  for (int kt = 0; kt < 8; ++kt) {
    int k0 = jbase + (kt << 5);
    v16h a = load_a_frag(&atile[0][0] + k0, NSEQ);
    #pragma unroll
    for (int dt = 0; dt < 4; ++dt) {
      v16h b = load_b_frag(vt + (size_t)(dt << 4) * NSEQ + k0, NSEQ);
      o[dt] = wmma_f16(a, b, o[dt]);
    }
  }
  #pragma unroll
  for (int dt = 0; dt < 4; ++dt) {
    #pragma unroll
    for (int r = 0; r < 8; ++r)
      osum[wave][(radd + r) * DHEAD + (dt << 4) + lcol] = o[dt][r];
  }
  __syncthreads();

  // ---- cross-wave reduce + store ----
  int b_ = bh >> 3, h_ = bh & 7;
  #pragma unroll
  for (int u = 0; u < 8; ++u) {
    int e = tid * 8 + u;
    float v = osum[0][e] + osum[1][e] + osum[2][e] + osum[3][e];
    int rr = e >> 6, d = e & 63;
    aout[((size_t)b_ * NSEQ + row0 + rr) * INNER + (h_ << 6) + d] = (h16)v;
  }
}

// ---------------------------------------------------------------------------
// Depthwise 3x3 conv (SAME) + bias + exact GELU; f32 in [b][n][96] -> f16 out
// ---------------------------------------------------------------------------
__global__ void __launch_bounds__(256) dwconv_gelu_kernel(const float* __restrict__ y1,
                                                          const float* __restrict__ w,
                                                          const float* __restrict__ bias,
                                                          h16* __restrict__ out) {
  int idx = blockIdx.x * blockDim.x + threadIdx.x;
  if (idx >= BB * NSEQ * HID) return;
  int c = idx % HID;
  int n = (idx / HID) & 1023;
  int b = idx / (HID * NSEQ);
  int i = n >> 5, j = n & 31;
  float acc = bias[c];
  #pragma unroll
  for (int ky = 0; ky < 3; ++ky) {
    int ii = i + ky - 1;
    if (ii < 0 || ii > 31) continue;
    #pragma unroll
    for (int kx = 0; kx < 3; ++kx) {
      int jj = j + kx - 1;
      if (jj < 0 || jj > 31) continue;
      acc += y1[((size_t)(b << 10) + (ii << 5) + jj) * HID + c] * w[c * 9 + ky * 3 + kx];
    }
  }
  float g = 0.5f * acc * (1.0f + erff(acc * 0.70710678118654752f));
  out[idx] = (h16)g;
}

// ---------------------------------------------------------------------------
// Host launcher
// ---------------------------------------------------------------------------
extern "C" void kernel_launch(void* const* d_in, const int* in_sizes, int n_in,
                              void* d_out, int out_size, void* d_ws, size_t ws_size,
                              hipStream_t stream) {
  (void)in_sizes; (void)n_in; (void)out_size; (void)ws_size;

  const float* x0    = (const float*)d_in[0];
  const float* ln1_w = (const float*)d_in[1];
  const float* ln1_b = (const float*)d_in[2];
  const float* w_qkv = (const float*)d_in[3];
  const float* w_out = (const float*)d_in[4];
  const float* b_out = (const float*)d_in[5];
  const float* ln2_w = (const float*)d_in[6];
  const float* ln2_b = (const float*)d_in[7];
  const float* fc1_w = (const float*)d_in[8];
  const float* fc1_b = (const float*)d_in[9];
  const float* dw_w  = (const float*)d_in[10];
  const float* dw_b  = (const float*)d_in[11];
  const float* fc2_w = (const float*)d_in[12];
  const float* fc2_b = (const float*)d_in[13];

  // ---- deterministic workspace carve-out ----
  char* p = (char*)d_ws;
  auto carve = [&](size_t bytes) -> char* {
    char* r = p;
    p += (bytes + 255) & ~(size_t)255;
    return r;
  };
  h16*   wqkv_t = (h16*)  carve((size_t)DEPTH * 1536 * DIM * 2);
  h16*   wout_t = (h16*)  carve((size_t)DEPTH * DIM * INNER * 2);
  h16*   wfc1_t = (h16*)  carve((size_t)DEPTH * HID * DIM * 2);
  h16*   wfc2_t = (h16*)  carve((size_t)DEPTH * DIM * HID * 2);
  h16*   xnh    = (h16*)  carve((size_t)NTOK * DIM * 2);
  h16*   qbuf   = (h16*)  carve((size_t)BB * HEADS * NSEQ * DHEAD * 2);
  h16*   kbuf   = (h16*)  carve((size_t)BB * HEADS * NSEQ * DHEAD * 2);
  h16*   vtbuf  = (h16*)  carve((size_t)BB * HEADS * DHEAD * NSEQ * 2);
  h16*   aout   = (h16*)  carve((size_t)NTOK * INNER * 2);
  float* y1     = (float*)carve((size_t)NTOK * HID * 4);
  h16*   yh     = (h16*)  carve((size_t)NTOK * HID * 2);
  float* xA     = (float*)carve((size_t)NTOK * DIM * 4);
  float* xB     = (float*)carve((size_t)NTOK * DIM * 4);
  h16*   attnl  = (h16*)  carve((size_t)BB * HEADS * NSEQ * NSEQ * 2); // 134 MB

  // ---- weight transposes (f32 -> f16 B^T) ----
  for (int l = 0; l < DEPTH; ++l) {
    transpose_f16_kernel<<<(DIM * 1536 + 255) / 256, 256, 0, stream>>>(
        w_qkv + (size_t)l * DIM * 1536, wqkv_t + (size_t)l * 1536 * DIM, DIM, 1536);
    transpose_f16_kernel<<<(INNER * DIM + 255) / 256, 256, 0, stream>>>(
        w_out + (size_t)l * INNER * DIM, wout_t + (size_t)l * DIM * INNER, INNER, DIM);
    transpose_f16_kernel<<<(DIM * HID + 255) / 256, 256, 0, stream>>>(
        fc1_w + (size_t)l * DIM * HID, wfc1_t + (size_t)l * HID * DIM, DIM, HID);
    transpose_f16_kernel<<<(HID * DIM + 255) / 256, 256, 0, stream>>>(
        fc2_w + (size_t)l * HID * DIM, wfc2_t + (size_t)l * DIM * HID, HID, DIM);
  }

  const float* cur = x0;
  for (int l = 0; l < DEPTH; ++l) {
    // PreNorm 1 -> f16
    ln_kernel<<<NTOK, 128, 0, stream>>>(cur, ln1_w + l * DIM, ln1_b + l * DIM, xnh);
    // QKV projection (64x64 tiles)
    gemm_qkv_kernel<<<dim3(1536 / 64, NTOK / 64), 32, 0, stream>>>(
        xnh, wqkv_t + (size_t)l * 1536 * DIM, qbuf, kbuf, vtbuf);
    // Attention with post-softmax cross-layer residual (f16 attn_last)
    attn_kernel<<<BB * HEADS * (NSEQ / 16), 128, 0, stream>>>(
        qbuf, kbuf, vtbuf, attnl, aout, l > 0 ? 1 : 0);
    // Out projection + bias + residual -> xA (64x64 tiles)
    gemm_kernel<64, 64><<<dim3(DIM / 64, NTOK / 64), 32, 0, stream>>>(
        aout, wout_t + (size_t)l * DIM * INNER, b_out + l * DIM, cur, xA,
        DIM, INNER, INNER, INNER, 1);
    // PreNorm 2 -> f16
    ln_kernel<<<NTOK, 128, 0, stream>>>(xA, ln2_w + l * DIM, ln2_b + l * DIM, xnh);
    // fc1 + bias -> y1 (f32)
    gemm_kernel<32, 32><<<dim3(HID / 32, NTOK / 32), 32, 0, stream>>>(
        xnh, wfc1_t + (size_t)l * HID * DIM, fc1_b + l * HID, nullptr, y1,
        HID, DIM, DIM, DIM, 0);
    // depthwise conv + bias + GELU -> f16
    dwconv_gelu_kernel<<<(BB * NSEQ * HID + 255) / 256, 256, 0, stream>>>(
        y1, dw_w + (size_t)l * HID * 9, dw_b + l * HID, yh);
    // fc2 + bias + residual(xA) -> next x (d_out on final layer)
    float* xnext = (l == DEPTH - 1) ? (float*)d_out : xB;
    gemm_kernel<32, 32><<<dim3(DIM / 32, NTOK / 32), 32, 0, stream>>>(
        yh, wfc2_t + (size_t)l * DIM * HID, fc2_b + l * DIM, xA, xnext,
        DIM, HID, HID, HID, 1);
    cur = xnext;
  }
}
